// WaveRNN_29059748725558
// MI455X (gfx1250) — matrix operand order repaired and checked by reading, not compiled, not run on hardware
//
#include <hip/hip_runtime.h>

typedef unsigned short ushort_t;
typedef __attribute__((ext_vector_type(16))) __bf16 v16bf;
typedef __attribute__((ext_vector_type(8)))  float  v8f;

#define SIZE_   896
#define HALF_   448
#define BINS_   256
#define B_      16
#define T_      1000
#define NTILE_HP 168      // 2688 / 16
#define KCH_    28        // 896 / 32
#define KCHH_   14        // 448 / 32
#define NBLK_   21        // 21 blocks * 8 waves = 168 hp tiles
#define ABYTES_ (KCH_ * 32 * 16 * 2)   // 28672 B, bf16 hidden in WMMA A layout

// ---- workspace layout (bytes, 256-aligned) ----
#define WS_BAR   0          // 2 ints: count, generation
#define WS_A     256        // 28*32*16 ushort  = 28672 B  (hidden, WMMA A layout, bf16)
#define WS_HP    28928      // 16*2688 f32      = 172032 B
#define WS_WH    200960     // 168*28*32*16 ushort = 4816896 B (W_h swizzled bf16)
#define WS_WBC   5017856    // 16*14*32*16 ushort  = 229376 B
#define WS_WBF   5247232    // 229376 B
#define WS_CH    5476608    // 16*448 f32 = 28672 B
#define WS_FH    5505280    // 28672 B

__device__ __forceinline__ ushort_t f2bf(float f) {
  unsigned u = __float_as_uint(f);
  return (ushort_t)((u + 0x7FFFu + ((u >> 16) & 1u)) >> 16);
}

__device__ __forceinline__ float sigmoidf_(float x) { return 1.0f / (1.0f + expf(-x)); }

__device__ __forceinline__ float gumbel_(unsigned long long x) {
  x += 0x9E3779B97F4A7C15ull;
  x = (x ^ (x >> 30)) * 0xBF58476D1CE4E5B9ull;
  x = (x ^ (x >> 27)) * 0x94D049BB133111EBull;
  x ^= (x >> 31);
  float u = ((float)(unsigned)(x >> 40) + 0.5f) * (1.0f / 16777216.0f); // (0,1)
  return -logf(-logf(u));
}

// async global -> LDS copy (CDNA5: tracked by ASYNCcnt).
// VDST operand = LDS byte offset; VADDR = 64-bit global address.
__device__ __forceinline__ void async_copy_b128(unsigned lds_off, const void* gptr) {
  asm volatile("global_load_async_to_lds_b128 %0, %1, off"
               :: "v"(lds_off), "v"(gptr) : "memory");
}
__device__ __forceinline__ void wait_asynccnt0() {
  asm volatile("s_wait_asynccnt 0x0" ::: "memory");
}

// grid-wide barrier: atomic arrive + generation spin (all NBLK_ blocks resident)
__device__ __forceinline__ void grid_barrier(int* bar) {
  __threadfence();
  __syncthreads();
  if (threadIdx.x == 0) {
    int g = __hip_atomic_load(bar + 1, __ATOMIC_RELAXED, __HIP_MEMORY_SCOPE_AGENT);
    if (atomicAdd(bar, 1) == NBLK_ - 1) {
      bar[0] = 0;
      __threadfence();
      atomicAdd(bar + 1, 1);
    } else {
      while (__hip_atomic_load(bar + 1, __ATOMIC_RELAXED, __HIP_MEMORY_SCOPE_AGENT) == g)
        __builtin_amdgcn_s_sleep(1);
    }
  }
  __syncthreads();
}

// ---- weight pre-swizzle into per-lane WMMA layouts (bf16, contiguous k = 16*(lane/16) + e) ----
// B-matrix 32x16 bf16: lane l (<16): N=l, K=0..15 ; lane l (>=16): N=l-16, K=16..31
__global__ void prep_wh_kernel(const float* __restrict__ W, ushort_t* __restrict__ out) {
  int idx = blockIdx.x * blockDim.x + threadIdx.x;
  if (idx >= NTILE_HP * KCH_ * 32 * 16) return;
  int e = idx & 15, l = (idx >> 4) & 31;
  int c = (idx >> 9) % KCH_;
  int n = (idx >> 9) / KCH_;
  int N = n * 16 + (l & 15);
  int K = c * 32 + ((l < 16) ? 0 : 16) + e;
  out[idx] = f2bf(W[N * SIZE_ + K]);
}

__global__ void prep_wbins_kernel(const float* __restrict__ W, ushort_t* __restrict__ out) {
  int idx = blockIdx.x * blockDim.x + threadIdx.x;
  if (idx >= 16 * KCHH_ * 32 * 16) return;
  int e = idx & 15, l = (idx >> 4) & 31;
  int c = (idx >> 9) % KCHH_;
  int n = (idx >> 9) / KCHH_;
  int N = n * 16 + (l & 15);
  int K = c * 32 + ((l < 16) ? 0 : 16) + e;
  out[idx] = f2bf(W[N * HALF_ + K]);
}

__global__ void init_bar_kernel(int* bar) {
  if (threadIdx.x < 2) bar[threadIdx.x] = 0;
}

// ---- persistent main kernel: all T steps in one launch ----
__global__ __launch_bounds__(256)
void wavernn_main(const float* __restrict__ cond, const float* __restrict__ gib,
                  const float* __restrict__ b_h,
                  const float* __restrict__ W_ci, const float* __restrict__ b_ci,
                  const float* __restrict__ W_fi, const float* __restrict__ b_fi,
                  const float* __restrict__ b_bc, const float* __restrict__ b_bf,
                  const ushort_t* __restrict__ WhS, const ushort_t* __restrict__ WbcS,
                  const ushort_t* __restrict__ WbfS,
                  ushort_t* __restrict__ Abuf, float* __restrict__ hp,
                  float* __restrict__ ch, float* __restrict__ fh,
                  int* bar, float* __restrict__ out) {
  __shared__ ushort_t s_A[KCH_ * 32 * 16];     // 28672 B: per-block LDS copy of hidden (bf16)
  __shared__ float s_logits[B_][BINS_];
  __shared__ float s_cv[B_], s_fv[B_], s_cvn[B_], s_fvn[B_];

  const int tid  = threadIdx.x;
  const int bid  = blockIdx.x;
  const int lane = tid & 31;
  const int wave = tid >> 5;
  const unsigned sA_base = (unsigned)(uintptr_t)(&s_A[0]);  // LDS aperture: low 32 bits = LDS offset

  if (bid == 0) {
    for (int i = tid; i < B_ * HALF_; i += 256) { ch[i] = 0.f; fh[i] = 0.f; }
    for (int i = tid; i < KCH_ * 32 * 16; i += 256) Abuf[i] = 0;  // bf16 zero
    if (tid < B_) { s_cv[tid] = -1.f; s_fv[tid] = -1.f; s_cvn[tid] = 0.f; s_fvn[tid] = 0.f; }
  }
  grid_barrier(bar);

  const int nt    = bid * 8 + wave;              // hp output tile 0..167
  const int col   = nt * 16 + (lane & 15);
  const float hbias = b_h[col];
  const int rbase = (lane >> 4) * 8;             // C/D layout: M = r + 8*(lane/16)

  for (int t = 0; t < T_; ++t) {
    // ---- stage hidden A fragments into LDS via async copy (ASYNCcnt) ----
    #pragma unroll
    for (int i = tid; i < ABYTES_ / 16; i += 256)
      async_copy_b128(sA_base + (unsigned)i * 16u, (const char*)Abuf + (size_t)i * 16u);
    wait_asynccnt0();
    __syncthreads();

    // ===== phase 1 (all blocks): hp = hidden @ W_h^T + b_h via v_wmma bf16 =====
    v8f acc = {};
    #pragma unroll 4
    for (int c = 0; c < KCH_; ++c) {
      v16bf a = *(const v16bf*)(s_A + (c * 32 + lane) * 16);                 // ds_load
      v16bf b = *(const v16bf*)(WhS + ((nt * KCH_ + c) * 32 + lane) * 16);   // L2-hot
      acc = __builtin_amdgcn_wmma_f32_16x16x32_bf16(false, a, false, b,
                                                    (short)0, acc, false, false);
    }
    #pragma unroll
    for (int r = 0; r < 8; ++r)
      hp[(rbase + r) * 2688 + col] = acc[r] + hbias;

    grid_barrier(bar);

    // ===== phase 2 (block 0): gates / state / bins-GEMM / sampling =====
    if (bid == 0) {
      for (int hh = 0; hh < 2; ++hh) {           // 0 = coarse, 1 = fine
        float* hs = hh ? fh : ch;
        const float* W_in = hh ? W_fi : W_ci;
        const float* b_in = hh ? b_fi : b_ci;
        const int nin = 2 + hh;

        // GRU gates + state update (7168 elems / 256 thr)
        for (int idx = tid; idx < B_ * HALF_; idx += 256) {
          const int b = idx / HALF_, i = idx - b * HALF_;
          const float* cb = cond + (size_t)(b * T_ + t) * 3 * SIZE_ + hh * HALF_ + i;
          float pr = cb[0]         + gib[0 * SIZE_ + hh * HALF_ + i];
          float pu = cb[SIZE_]     + gib[1 * SIZE_ + hh * HALF_ + i];
          float pe = cb[2 * SIZE_] + gib[2 * SIZE_ + hh * HALF_ + i];
          const float in0 = s_cv[b], in1 = s_fv[b], in2 = s_cvn[b];
          const int jr = i, ju = HALF_ + i, je = 2 * HALF_ + i;
          pr += b_in[jr] + in0 * W_in[jr * nin] + in1 * W_in[jr * nin + 1]
                         + (hh ? in2 * W_in[jr * nin + 2] : 0.f);
          pu += b_in[ju] + in0 * W_in[ju * nin] + in1 * W_in[ju * nin + 1]
                         + (hh ? in2 * W_in[ju * nin + 2] : 0.f);
          pe += b_in[je] + in0 * W_in[je * nin] + in1 * W_in[je * nin + 1]
                         + (hh ? in2 * W_in[je * nin + 2] : 0.f);
          const float* hpb = hp + b * 2688;
          float r = sigmoidf_(pr + hpb[hh * HALF_ + i]);
          float u = sigmoidf_(pu + hpb[896 + hh * HALF_ + i]);
          float e = tanhf(r * hpb[1792 + hh * HALF_ + i] + pe);
          hs[idx] = u * hs[idx] + (1.f - u) * e;
        }
        __syncthreads();

        // repack new half-state into WMMA A operand bf16 (A 16x32, contiguous k = 16*(l/16) + e, as for B)
        for (int idx = tid; idx < KCHH_ * 32 * 16; idx += 256) {
          int e = idx & 15, l = (idx >> 4) & 31, c = idx >> 9;
          int K = c * 32 + ((l < 16) ? 0 : 16) + e;
          Abuf[((hh * KCHH_ + c) * 32 + l) * 16 + e] = f2bf(hs[(l & 15) * HALF_ + K]);
        }
        __syncthreads();

        // logits = hs @ W_bins^T + b_bins (16 N-tiles x 14 K-chunks, WMMA)
        const ushort_t* WbS = hh ? WbfS : WbcS;
        const float* bb = hh ? b_bf : b_bc;
        for (int tt = wave; tt < 16; tt += 8) {
          v8f accl = {};
          #pragma unroll 2
          for (int c = 0; c < KCHH_; ++c) {
            v16bf a  = *(const v16bf*)(Abuf + ((hh * KCHH_ + c) * 32 + lane) * 16);
            v16bf bm = *(const v16bf*)(WbS + ((tt * KCHH_ + c) * 32 + lane) * 16);
            accl = __builtin_amdgcn_wmma_f32_16x16x32_bf16(false, a, false, bm,
                                                           (short)0, accl, false, false);
          }
          float bbv = bb[tt * 16 + (lane & 15)];
          #pragma unroll
          for (int r = 0; r < 8; ++r)
            s_logits[rbase + r][tt * 16 + (lane & 15)] = accl[r] + bbv;
        }
        __syncthreads();

        // Gumbel-max sampling == categorical(log_softmax(logits))
        for (int row = wave; row < 16; row += 8) {
          float best = -3.4e38f; int bi = 0;
          #pragma unroll
          for (int k = 0; k < 8; ++k) {
            int bin = lane + k * 32;
            unsigned long long ctr =
                ((((unsigned long long)(t * 2 + hh)) * 16ull + (unsigned)row) << 8) | (unsigned)bin;
            float v = s_logits[row][bin] + gumbel_(ctr);
            if (v > best) { best = v; bi = bin; }
          }
          for (int off = 16; off >= 1; off >>= 1) {
            float ob = __shfl_xor(best, off, 32);
            int   oi = __shfl_xor(bi, off, 32);
            if (ob > best || (ob == best && oi < bi)) { best = ob; bi = oi; }
          }
          if (lane == 0) {
            float sv = (float)bi * (1.0f / 127.5f) - 1.0f;
            if (hh) s_fvn[row] = sv; else s_cvn[row] = sv;
            out[hh * (B_ * T_) + row * T_ + t] = (float)bi;
          }
        }
        __syncthreads();
      }
      if (tid < B_) { s_cv[tid] = s_cvn[tid]; s_fv[tid] = s_fvn[tid]; }
      __syncthreads();
    }
    grid_barrier(bar);   // publishes new Abuf (bf16 hidden) to all blocks
  }

  if (bid == 0) {        // final hidden = concat(ch, fh)
    for (int i = tid; i < B_ * SIZE_; i += 256) {
      int b = i / SIZE_, k = i - b * SIZE_;
      out[2 * B_ * T_ + i] = (k < HALF_) ? ch[b * HALF_ + k] : fh[b * HALF_ + k - HALF_];
    }
  }
}

extern "C" void kernel_launch(void* const* d_in, const int* in_sizes, int n_in,
                              void* d_out, int out_size, void* d_ws, size_t ws_size,
                              hipStream_t stream) {
  const float* cond = (const float*)d_in[0];
  const float* gib  = (const float*)d_in[1];
  const float* W_h  = (const float*)d_in[2];
  const float* b_h  = (const float*)d_in[3];
  const float* W_ci = (const float*)d_in[4];
  const float* b_ci = (const float*)d_in[5];
  const float* W_fi = (const float*)d_in[6];
  const float* b_fi = (const float*)d_in[7];
  const float* W_bc = (const float*)d_in[8];
  const float* b_bc = (const float*)d_in[9];
  const float* W_bf = (const float*)d_in[10];
  const float* b_bf = (const float*)d_in[11];

  char* ws = (char*)d_ws;
  int*      bar  = (int*)(ws + WS_BAR);
  ushort_t* Abuf = (ushort_t*)(ws + WS_A);
  float*    hp   = (float*)(ws + WS_HP);
  ushort_t* WhS  = (ushort_t*)(ws + WS_WH);
  ushort_t* WbcS = (ushort_t*)(ws + WS_WBC);
  ushort_t* WbfS = (ushort_t*)(ws + WS_WBF);
  float*    ch   = (float*)(ws + WS_CH);
  float*    fh   = (float*)(ws + WS_FH);

  init_bar_kernel<<<1, 64, 0, stream>>>(bar);
  const int totWh = NTILE_HP * KCH_ * 32 * 16;
  prep_wh_kernel<<<(totWh + 255) / 256, 256, 0, stream>>>(W_h, WhS);
  const int totWb = 16 * KCHH_ * 32 * 16;
  prep_wbins_kernel<<<(totWb + 255) / 256, 256, 0, stream>>>(W_bc, WbcS);
  prep_wbins_kernel<<<(totWb + 255) / 256, 256, 0, stream>>>(W_bf, WbfS);

  wavernn_main<<<NBLK_, 256, 0, stream>>>(cond, gib, b_h, W_ci, b_ci, W_fi, b_fi,
                                          b_bc, b_bf, WhS, WbcS, WbfS,
                                          Abuf, hp, ch, fh, bar, (float*)d_out);
}
